// qlinear_8bit_50629074485862
// MI455X (gfx1250) — 8-bit matrix operand order repaired and checked by reading, compiled, not run on hardware
//
#include <hip/hip_runtime.h>
#include <hip/hip_fp16.h>

typedef int v8i __attribute__((ext_vector_type(8)));

#define M_DIM 32768
#define K_DIM 2048
#define N_DIM 2048
#define KT    (K_DIM / 64)
#define A_TILES_BYTES ((size_t)(M_DIM / 16) * KT * 1024)

__device__ __forceinline__ int quant8(float xf, float inv_delta, float zp) {
    float v = rintf(fmaf(xf, inv_delta, zp));
    v = fminf(fmaxf(v, -128.0f), 127.0f);
    return (int)v;
}

__global__ __launch_bounds__(256) void quant_swizzle_kernel(
    const __half* __restrict__ x,
    const __half* __restrict__ act_delta,
    const __half* __restrict__ act_zp,
    signed char*  __restrict__ qs)
{
    const float inv_delta = 1.0f / __half2float(act_delta[0]);
    const float zp        = __half2float(act_zp[0]);

    const int lane = threadIdx.x & 31;
    const int wave = threadIdx.x >> 5;
    const int tile = blockIdx.x * 8 + wave;
    const int mt   = tile / KT;
    const int kt   = tile % KT;
    const int m    = mt * 16 + (lane & 15);
    const int h    = lane >> 4;

    const __half* src = x + (size_t)m * K_DIM + kt * 64 + h * 16;

    int outw[8];
#pragma unroll
    for (int c = 0; c < 4; ++c) {
        union { int4 i4; __half2 h2[4]; } in;
        in.i4 = *(const int4*)(src + (c & 1) * 8 + (c >> 1) * 32);
        int b[8];
#pragma unroll
        for (int i = 0; i < 4; ++i) {
            float2 f = __half22float2(in.h2[i]);
            b[2 * i + 0] = quant8(f.x, inv_delta, zp);
            b[2 * i + 1] = quant8(f.y, inv_delta, zp);
        }
        outw[2 * c + 0] = (b[0] & 0xff) | ((b[1] & 0xff) << 8) |
                          ((b[2] & 0xff) << 16) | (b[3] << 24);
        outw[2 * c + 1] = (b[4] & 0xff) | ((b[5] & 0xff) << 8) |
                          ((b[6] & 0xff) << 16) | (b[7] << 24);
    }

    int4* dst = (int4*)(qs + (size_t)tile * 1024 + lane * 32);
    dst[0] = make_int4(outw[0], outw[1], outw[2], outw[3]);
    dst[1] = make_int4(outw[4], outw[5], outw[6], outw[7]);
}

__global__ __launch_bounds__(256) void w_swizzle_kernel(
    const signed char* __restrict__ w,
    signed char*       __restrict__ wsw)
{
    const int lane = threadIdx.x & 31;
    const int wave = threadIdx.x >> 5;
    const int tile = blockIdx.x * 8 + wave;
    const int nt   = tile / KT;
    const int kt   = tile % KT;
    const int n    = nt * 16 + (lane & 15);
    const int h    = lane >> 4;

    const signed char* src = w + (size_t)n * K_DIM + kt * 64 + h * 16;
    int4 g0 = *(const int4*)(src);
    int4 g1 = *(const int4*)(src + 32);

    int4* dst = (int4*)(wsw + (size_t)tile * 1024 + lane * 32);
    dst[0] = g0;
    dst[1] = g1;
}

__global__ __launch_bounds__(256) void wmma_iu8_gemm_kernel(
    const signed char* __restrict__ qs,
    const signed char* __restrict__ wsw,
    const float* __restrict__ atwd,
    const float* __restrict__ zpsum,
    const float* __restrict__ bias,
    __half* __restrict__ out)
{
    const int lane = threadIdx.x & 31;
    const int wave = threadIdx.x >> 5;
    const int l16  = lane & 15;
    const int half = lane >> 4;

    const int waveM = wave & 1;
    const int waveN = wave >> 1;

    const int mtile0 = blockIdx.y * 8 + waveM * 4;
    const int ntile0 = blockIdx.x * 8 + waveN * 2;

    v8i acc[4][2];
    const v8i vzero = {0, 0, 0, 0, 0, 0, 0, 0};
#pragma unroll
    for (int mt = 0; mt < 4; ++mt)
#pragma unroll
        for (int nt = 0; nt < 2; ++nt)
            acc[mt][nt] = vzero;

    const signed char* aBase[4];
#pragma unroll
    for (int mt = 0; mt < 4; ++mt)
        aBase[mt] = qs + (size_t)(mtile0 + mt) * KT * 1024 + lane * 32;
    const signed char* bBase[2];
#pragma unroll
    for (int nt = 0; nt < 2; ++nt)
        bBase[nt] = wsw + (size_t)(ntile0 + nt) * KT * 1024 + lane * 32;

    union Frag { v8i v; int4 d[2]; };

    for (int kt = 0; kt < KT; ++kt) {
        const int koff = kt << 10;
        Frag a[4];
#pragma unroll
        for (int mt = 0; mt < 4; ++mt) {
            a[mt].d[0] = *(const int4*)(aBase[mt] + koff);
            a[mt].d[1] = *(const int4*)(aBase[mt] + koff + 16);
        }
        Frag b[2];
#pragma unroll
        for (int nt = 0; nt < 2; ++nt) {
            b[nt].d[0] = *(const int4*)(bBase[nt] + koff);
            b[nt].d[1] = *(const int4*)(bBase[nt] + koff + 16);
        }
#pragma unroll
        for (int mt = 0; mt < 4; ++mt) {
#pragma unroll
            for (int nt = 0; nt < 2; ++nt) {
                acc[mt][nt] = __builtin_amdgcn_wmma_i32_16x16x64_iu8(
                    true, a[mt].v, true, b[nt].v, acc[mt][nt], false, false);
            }
        }
    }

    const int mBase = mtile0 * 16;
    const int nBase = ntile0 * 16;
#pragma unroll
    for (int nt = 0; nt < 2; ++nt) {
        const int col   = nBase + nt * 16 + l16;
        const float scl = atwd[col];
        const float off = bias[col] - zpsum[col];
#pragma unroll
        for (int mt = 0; mt < 4; ++mt) {
#pragma unroll
            for (int r = 0; r < 8; ++r) {
                const int row = mBase + mt * 16 + r + half * 8;
                const float v = fmaf((float)acc[mt][nt][r], scl, off);
                __builtin_nontemporal_store(
                    __half_as_ushort(__float2half(v)),
                    (unsigned short*)(out + (size_t)row * N_DIM + col));
            }
        }
    }
}

extern "C" void kernel_launch(void* const* d_in, const int* in_sizes, int n_in,
                              void* d_out, int out_size, void* d_ws, size_t ws_size,
                              hipStream_t stream)
{
    const __half*      x         = (const __half*)d_in[0];
    const __half*      act_delta = (const __half*)d_in[1];
    const __half*      act_zp    = (const __half*)d_in[2];
    const float*       zpsum     = (const float*)d_in[3];
    const float*       atwd      = (const float*)d_in[4];
    const float*       bias      = (const float*)d_in[5];
    const signed char* wq        = (const signed char*)d_in[6];
    __half*            out       = (__half*)d_out;

    signed char* qs  = (signed char*)d_ws;
    signed char* wsw = (signed char*)d_ws + A_TILES_BYTES;

    quant_swizzle_kernel<<<(M_DIM / 16) * KT / 8, 256, 0, stream>>>(
        x, act_delta, act_zp, qs);

    w_swizzle_kernel<<<(N_DIM / 16) * KT / 8, 256, 0, stream>>>(wq, wsw);

    dim3 grid(N_DIM / 128, M_DIM / 128);
    wmma_iu8_gemm_kernel<<<grid, 256, 0, stream>>>(qs, wsw, atwd, zpsum, bias, out);
}
